// SVM_He_996432413412
// MI455X (gfx1250) — compile-verified, matrix operand order checked by reading, not run on hardware
//
#include <hip/hip_runtime.h>
#include <hip/hip_bf16.h>

typedef __attribute__((ext_vector_type(16))) __bf16 v16bf;
typedef __attribute__((ext_vector_type(8)))  float  v8f;

#define NWAY     10
#define KSHOT    5
#define QSHOT    15
#define DDIM     1024
#define NSUP     50     // NWAY*KSHOT
#define NQRY     150    // NWAY*QSHOT
#define GS_LD    65     // padded LDS stride for 64x64 G (bank-conflict free)
#define POW_IT   30
#define QP_IT    300
#define C_REG    0.01f

// Build one 16-element bf16 WMMA fragment (A or B) for v_wmma_f32_16x16x32_bf16.
// 16-bit operand layout (ISA 7.12.2): lanes 0-15 hold M(or N)=lane rows;
// half-wave hi=0 -> K = k0+[0..7] and k0+[16..23]; hi=1 -> K = k0+[8..15] and k0+[24..31].
// Branch-free: caller clamps the row index; garbage rows/cols are masked at the
// store/accumulate stage, so no zero-fill or EXEC games are needed in the hot loop.
__device__ __forceinline__ v16bf load_frag_bf16(const float* __restrict__ p) {
    v16bf r;
#pragma unroll
    for (int i = 0; i < 8; ++i) {
        r[i]     = (__bf16)p[i];
        r[i + 8] = (__bf16)p[i + 16];
    }
    return r;
}

__global__ __launch_bounds__(512)
void svm_he_qp_kernel(const float* __restrict__ support,   // [B,50,1024]
                      const float* __restrict__ query,     // [B,150,1024]
                      const int*   __restrict__ labels,    // [B,50]
                      float*       __restrict__ out) {     // [B,150,10]
    __shared__ float Gs[64 * GS_LD];       // padded 64x64 G (rows/cols >=50 are zero)
    __shared__ float outAcc[160 * NWAY];   // logits accumulator (150x10 used)
    __shared__ float vvec[64], yvec[64], avec[64], apvec[64], red[64];
    __shared__ int   lab[64];
    __shared__ float scal[2];              // [0]=norm tmp, [1]=step

    const int b   = blockIdx.x;
    const int tid = threadIdx.x;
    const int w   = tid >> 5;              // wave id (0..15)
    const int l   = tid & 31;              // lane id
    const int ln  = l & 15;
    const int hi  = l >> 4;

    const float* sb = support + (size_t)b * NSUP * DDIM;
    const float* qb = query   + (size_t)b * NQRY * DDIM;

    if (tid < NSUP) lab[tid] = labels[b * NSUP + tid];
    for (int idx = tid; idx < 160 * NWAY; idx += 512) outAcc[idx] = 0.0f;
    __syncthreads();

    // ---------------- Phase 1: G = (s s^T) .* G_lab into LDS (bf16 WMMA) -----------
    {
        const int tr = w >> 2, tc = w & 3;             // 4x4 tiles of 16 -> 64x64
        const int arow = tr * 16 + ln;                 // A row (support row)
        const int brow = tc * 16 + ln;                 // B "col" = support row
        // clamp OOB rows to a valid row; results masked at store
        const float* ap = sb + (size_t)min(arow, NSUP - 1) * DDIM + hi * 8;
        const float* bp = sb + (size_t)min(brow, NSUP - 1) * DDIM + hi * 8;
        v8f acc;
#pragma unroll
        for (int e = 0; e < 8; ++e) acc[e] = 0.0f;
#pragma unroll 2
        for (int k0 = 0; k0 < DDIM; k0 += 32) {
            v16bf af = load_frag_bf16(ap + k0);
            v16bf bf = load_frag_bf16(bp + k0);
            acc = __builtin_amdgcn_wmma_f32_16x16x32_bf16(false, af, false, bf,
                                                          (short)0, acc, false, false);
        }
        // C layout: VGPR e -> M = e (lanes 0-15) or 8+e (lanes 16-31); N = ln
#pragma unroll
        for (int e = 0; e < 8; ++e) {
            int row = tr * 16 + e + hi * 8;
            int col = tc * 16 + ln;
            float g = 0.0f;
            if (row < NSUP && col < NSUP) {
                float glab = (lab[row] == lab[col]) ? (10.0f / 9.0f) : (-10.0f / 81.0f);
                g = acc[e] * glab;
            }
            Gs[row * GS_LD + col] = g;
        }
    }
    __syncthreads();

    // ---------------- Phase 2: power iteration + FISTA box QP (fp32, LDS) ----------
    if (tid < 64) vvec[tid] = (tid < NSUP) ? 1.0f : 0.0f;
    __syncthreads();

    for (int it = 0; it < POW_IT; ++it) {
        float wv = 0.0f;
        if (tid < 64) {
            float s = 0.0f;
            for (int j = 0; j < NSUP; ++j) s += Gs[tid * GS_LD + j] * vvec[j];
            wv = s;
            red[tid] = s * s;
        }
        __syncthreads();
        if (tid == 0) {
            float s = 0.0f;
            for (int j = 0; j < 64; ++j) s += red[j];
            scal[0] = sqrtf(s);
        }
        __syncthreads();
        if (tid < 64) vvec[tid] = wv / (scal[0] + 1e-12f);
        __syncthreads();
    }
    if (tid < 64) {
        float s = 0.0f;
        for (int j = 0; j < NSUP; ++j) s += Gs[tid * GS_LD + j] * vvec[j];
        red[tid] = s * s;
    }
    __syncthreads();
    if (tid == 0) {
        float s = 0.0f;
        for (int j = 0; j < 64; ++j) s += red[j];
        scal[1] = 1.0f / (sqrtf(s) + 1e-6f);           // step = 1/(L+1e-6)
    }
    __syncthreads();

    if (tid < 64) { avec[tid] = 0.0f; apvec[tid] = 0.0f; }
    __syncthreads();
    float t_f = 1.0f;
    for (int it = 0; it < QP_IT; ++it) {
        if (tid < 64) {
            float co = (t_f - 1.0f) / (t_f + 2.0f);
            yvec[tid] = avec[tid] + co * (avec[tid] - apvec[tid]);
        }
        __syncthreads();
        float anew = 0.0f;
        if (tid < 64) {
            float g = 0.0f;
            for (int j = 0; j < NSUP; ++j) g += Gs[tid * GS_LD + j] * yvec[j];
            g -= 1.0f;                                  // e = -1
            anew = yvec[tid] - scal[1] * g;
            anew = fminf(fmaxf(anew, 0.0f), C_REG);
        }
        __syncthreads();
        if (tid < 64) { apvec[tid] = avec[tid]; avec[tid] = anew; }
        t_f += 1.0f;
        __syncthreads();
    }

    // ---------------- Phase 3: compat = q s^T, weight by qp, K-major fold ----------
    for (int t = w; t < 40; t += 16) {                  // 10 row-tiles x 4 col-tiles
        const int tr = t >> 2, tc = t & 3;
        const int arow = tr * 16 + ln;                  // query row
        const int brow = tc * 16 + ln;                  // support row (compat col)
        const float* ap = qb + (size_t)min(arow, NQRY - 1) * DDIM + hi * 8;
        const float* bp = sb + (size_t)min(brow, NSUP - 1) * DDIM + hi * 8;
        v8f acc;
#pragma unroll
        for (int e = 0; e < 8; ++e) acc[e] = 0.0f;
#pragma unroll 2
        for (int k0 = 0; k0 < DDIM; k0 += 32) {
            v16bf af = load_frag_bf16(ap + k0);
            v16bf bf = load_frag_bf16(bp + k0);
            acc = __builtin_amdgcn_wmma_f32_16x16x32_bf16(false, af, false, bf,
                                                          (short)0, acc, false, false);
        }
#pragma unroll
        for (int e = 0; e < 8; ++e) {
            int row = tr * 16 + e + hi * 8;             // query index
            int col = tc * 16 + ln;                     // support index = k*10 + n
            if (row < NQRY && col < NSUP) {
                atomicAdd(&outAcc[row * NWAY + (col % NWAY)], avec[col] * acc[e]);
            }
        }
    }
    __syncthreads();

    float* ob = out + (size_t)b * NQRY * NWAY;
    for (int idx = tid; idx < NQRY * NWAY; idx += 512) ob[idx] = outAcc[idx];
}

extern "C" void kernel_launch(void* const* d_in, const int* in_sizes, int n_in,
                              void* d_out, int out_size, void* d_ws, size_t ws_size,
                              hipStream_t stream) {
    (void)n_in; (void)d_ws; (void)ws_size; (void)out_size;
    const float* support = (const float*)d_in[0];
    const float* query   = (const float*)d_in[1];
    const int*   labels  = (const int*)d_in[2];
    float*       out     = (float*)d_out;
    const int B = in_sizes[0] / (NSUP * DDIM);          // 512
    svm_he_qp_kernel<<<B, 512, 0, stream>>>(support, query, labels, out);
}
